// GraphSAGE_28759101014107
// MI455X (gfx1250) — matrix operand order repaired and checked by reading, not compiled, not run on hardware
//
#include <hip/hip_runtime.h>
#include <hip/hip_bf16.h>

#define NODES 50000
#define EDGES 1600000
#define DIN   128
#define HID   128
#define DOUTC 64
#define EPSV  1e-5f
#define BN_ROWS 256

typedef __attribute__((ext_vector_type(16))) __bf16 v16bf;
typedef __attribute__((ext_vector_type(8)))  float  v8f;

// ---------------- utility ----------------
__global__ void k_zero(float* __restrict__ p, int n) {
    int i = blockIdx.x * blockDim.x + threadIdx.x;
    if (i < n) p[i] = 0.0f;
}

// ---------------- degree: deg[dst] += 1 ----------------
__global__ void k_degree(const int* __restrict__ dst, float* __restrict__ deg, int E) {
    int e = blockIdx.x * blockDim.x + threadIdx.x;
    if (e < E) unsafeAtomicAdd(&deg[dst[e]], 1.0f);
}

// ---------------- edge aggregation: agg[dst] += X[src] ----------------
// One wave per edge: lane handles 4 contiguous channels (b128 load, 4 f32 atomics).
// X and agg are L2-resident (25.6MB each), so this runs at L2 bandwidth.
__global__ void k_edge_agg(const float* __restrict__ X, const int* __restrict__ src,
                           const int* __restrict__ dst, float* __restrict__ agg, int E) {
    long long gid = (long long)blockIdx.x * blockDim.x + threadIdx.x;
    long long total = (long long)E * 32;
    if (gid >= total) return;
    int e = (int)(gid >> 5);
    int c = ((int)gid & 31) * 4;
    int s = src[e], d = dst[e];
    const float4 v = *(const float4*)(X + (size_t)s * DIN + c);
    float* o = agg + (size_t)d * DIN + c;
    unsafeAtomicAdd(o + 0, v.x);
    unsafeAtomicAdd(o + 1, v.y);
    unsafeAtomicAdd(o + 2, v.z);
    unsafeAtomicAdd(o + 3, v.w);
}

// ---------------- scatter-mean finalize: agg /= max(deg,1) ----------------
__global__ void k_divdeg(float* __restrict__ agg, const float* __restrict__ deg, int n) {
    int i = blockIdx.x * blockDim.x + threadIdx.x;
    if (i >= n) return;
    int r = i >> 7;  // / DIN (128)
    agg[i] = agg[i] / fmaxf(deg[r], 1.0f);
}

// pack 8 floats (two b128 payloads) into fragment elements [base, base+8)
__device__ __forceinline__ void pack8(v16bf& v, int base, float4 lo, float4 hi) {
    v[base + 0] = (__bf16)lo.x; v[base + 1] = (__bf16)lo.y;
    v[base + 2] = (__bf16)lo.z; v[base + 3] = (__bf16)lo.w;
    v[base + 4] = (__bf16)hi.x; v[base + 5] = (__bf16)hi.y;
    v[base + 6] = (__bf16)hi.z; v[base + 7] = (__bf16)hi.w;
}

// ---------------- fused SAGE GEMM via WMMA ----------------
// Y[m,n] = sum_k A1[m,k]*W1[n,k] + sum_k A2[m,k]*W2[n,k] + bias[n]
// One wave computes a 16x16 output tile; K=DIN=128 fully unrolled in 4 steps of 32
// (bf16 WMMA, f32 accumulate). Fragment layouts used here (wave32, contiguous k order, not the ISA table order):
//   A (16x32): lane half 0 holds K 0-15, half 1 holds K 16-31             -> 1x contiguous-16
//   B (32x16): lane half 0 holds K 0-15, half 1 holds K 16-31             -> 1x contiguous-16
// All fragment traffic is global_load_b128.
template <int DOUT_T>
__global__ void k_sage_gemm(const float* __restrict__ A1, const float* __restrict__ A2,
                            const float* __restrict__ W1, const float* __restrict__ W2,
                            const float* __restrict__ bias, float* __restrict__ Y,
                            int nrows) {
    constexpr int din    = DIN;
    constexpr int tilesN = DOUT_T / 16;     // power of two -> shifts, no div/mod
    int wave = (blockIdx.x * blockDim.x + threadIdx.x) >> 5;  // uniform per wave
    int lane = threadIdx.x & 31;
    int totalTiles = (nrows >> 4) * tilesN;  // nrows is a multiple of 16
    if (wave >= totalTiles) return;          // whole-wave early-out (EXEC stays all-1s)
    int tileM = wave / tilesN, tileN = wave % tilesN;
    int row0 = tileM << 4, col0 = tileN << 4;
    int half = lane >> 4, mr = lane & 15;
    int arow = row0 + mr;   // A-matrix row owned by this lane
    int bcol = col0 + mr;   // B-matrix column (== weight row) owned by this lane

    const float* pa1 = A1 + (size_t)arow * din;
    const float* pa2 = A2 + (size_t)arow * din;
    const float* pw1 = W1 + (size_t)bcol * din;
    const float* pw2 = W2 + (size_t)bcol * din;

    v8f acc = {};
#pragma unroll
    for (int k0 = 0; k0 < din; k0 += 32) {
        const int ab = k0 + half * 16;   // A fragment K base for this lane half
        const int bb = k0 + half * 16;   // B fragment K base for this lane half

        v16bf a1, a2, b1, b2;
        pack8(a1, 0, *(const float4*)(pa1 + ab),      *(const float4*)(pa1 + ab + 4));
        pack8(a1, 8, *(const float4*)(pa1 + ab + 8),  *(const float4*)(pa1 + ab + 12));
        pack8(a2, 0, *(const float4*)(pa2 + ab),      *(const float4*)(pa2 + ab + 4));
        pack8(a2, 8, *(const float4*)(pa2 + ab + 8),  *(const float4*)(pa2 + ab + 12));
        pack8(b1, 0, *(const float4*)(pw1 + bb),      *(const float4*)(pw1 + bb + 4));
        pack8(b1, 8, *(const float4*)(pw1 + bb + 8),  *(const float4*)(pw1 + bb + 12));
        pack8(b2, 0, *(const float4*)(pw2 + bb),      *(const float4*)(pw2 + bb + 4));
        pack8(b2, 8, *(const float4*)(pw2 + bb + 8),  *(const float4*)(pw2 + bb + 12));

        acc = __builtin_amdgcn_wmma_f32_16x16x32_bf16(false, a1, false, b1,
                                                      (short)0, acc, false, false);
        acc = __builtin_amdgcn_wmma_f32_16x16x32_bf16(false, a2, false, b2,
                                                      (short)0, acc, false, false);
    }

    // D layout: VGPR r -> M = 8*half + r, N = lane%16. Row stride is compile-time.
    float bv = bias[bcol];
    float* yp = Y + (size_t)(row0 + half * 8) * DOUT_T + bcol;
#pragma unroll
    for (int r = 0; r < 8; ++r) {
        yp[(size_t)r * DOUT_T] = acc[r] + bv;
    }
}

// ---------------- BatchNorm statistics: per-channel sum & sumsq ----------------
// blockDim.x == HID; each block reduces BN_ROWS rows locally, then 2 atomics/channel.
__global__ void k_bn_stats(const float* __restrict__ Y, float* __restrict__ stat, int nrows) {
    int c  = threadIdx.x;
    int r0 = blockIdx.x * BN_ROWS;
    int r1 = min(r0 + BN_ROWS, nrows);
    float s = 0.0f, s2 = 0.0f;
    for (int r = r0; r < r1; ++r) {
        float v = Y[(size_t)r * HID + c];
        s += v; s2 += v * v;
    }
    unsafeAtomicAdd(&stat[c],       s);
    unsafeAtomicAdd(&stat[HID + c], s2);
}

// ---------------- BN apply + ReLU (in place) ----------------
__global__ void k_bn_relu(float* __restrict__ Y, const float* __restrict__ stat,
                          const float* __restrict__ g, const float* __restrict__ be, int n) {
    int i = blockIdx.x * blockDim.x + threadIdx.x;
    if (i >= n) return;
    int c = i & (HID - 1);
    const float invN = 1.0f / (float)NODES;
    float mu  = stat[c] * invN;
    float var = stat[HID + c] * invN - mu * mu;
    float rs  = rsqrtf(var + EPSV);
    float v   = (Y[i] - mu) * rs * g[c] + be[c];
    Y[i] = fmaxf(v, 0.0f);
}

// ---------------- host orchestration ----------------
extern "C" void kernel_launch(void* const* d_in, const int* in_sizes, int n_in,
                              void* d_out, int out_size, void* d_ws, size_t ws_size,
                              hipStream_t stream) {
    const float* x   = (const float*)d_in[0];
    const int*   ei  = (const int*)d_in[1];
    const int*   src = ei;
    const int*   dst = ei + EDGES;
    const float *Wl0 = (const float*)d_in[2],  *Wr0 = (const float*)d_in[3],
                *b0  = (const float*)d_in[4],  *g0  = (const float*)d_in[5],
                *be0 = (const float*)d_in[6],
                *Wl1 = (const float*)d_in[7],  *Wr1 = (const float*)d_in[8],
                *b1  = (const float*)d_in[9],  *g1  = (const float*)d_in[10],
                *be1 = (const float*)d_in[11],
                *Wl2 = (const float*)d_in[12], *Wr2 = (const float*)d_in[13],
                *b2  = (const float*)d_in[14];
    float* out = (float*)d_out;

    float* ws   = (float*)d_ws;
    float* deg  = ws;                               // NODES
    float* agg  = deg  + NODES;                     // NODES*HID
    float* bufA = agg  + (size_t)NODES * HID;       // NODES*HID
    float* bufB = bufA + (size_t)NODES * HID;       // NODES*HID
    float* stat = bufB + (size_t)NODES * HID;       // 2*HID

    const long long nagg = (long long)NODES * HID;
    const long long eth  = (long long)EDGES * 32;
    const int zb  = 256;
    const int gemmBlocksHid = (int)(((NODES / 16) * (HID / 16)   + 7) / 8);  // 8 waves/block
    const int gemmBlocksOut = (int)(((NODES / 16) * (DOUTC / 16) + 7) / 8);
    const int bnBlocks = (NODES + BN_ROWS - 1) / BN_ROWS;

    // degree (shared by all layers)
    k_zero<<<(NODES + zb - 1) / zb, zb, 0, stream>>>(deg, NODES);
    k_degree<<<(EDGES + zb - 1) / zb, zb, 0, stream>>>(dst, deg, EDGES);

    // ---- layer 0: SAGE(IN->H) + BN + ReLU  -> bufA ----
    k_zero<<<(int)((nagg + zb - 1) / zb), zb, 0, stream>>>(agg, (int)nagg);
    k_edge_agg<<<(int)((eth + zb - 1) / zb), zb, 0, stream>>>(x, src, dst, agg, EDGES);
    k_divdeg<<<(int)((nagg + zb - 1) / zb), zb, 0, stream>>>(agg, deg, (int)nagg);
    k_sage_gemm<HID><<<gemmBlocksHid, 256, 0, stream>>>(agg, x, Wl0, Wr0, b0, bufA, NODES);
    k_zero<<<1, 2 * HID, 0, stream>>>(stat, 2 * HID);
    k_bn_stats<<<bnBlocks, HID, 0, stream>>>(bufA, stat, NODES);
    k_bn_relu<<<(int)((nagg + zb - 1) / zb), zb, 0, stream>>>(bufA, stat, g0, be0, (int)nagg);

    // ---- layer 1: SAGE(H->H) + BN + ReLU  -> bufB ----
    k_zero<<<(int)((nagg + zb - 1) / zb), zb, 0, stream>>>(agg, (int)nagg);
    k_edge_agg<<<(int)((eth + zb - 1) / zb), zb, 0, stream>>>(bufA, src, dst, agg, EDGES);
    k_divdeg<<<(int)((nagg + zb - 1) / zb), zb, 0, stream>>>(agg, deg, (int)nagg);
    k_sage_gemm<HID><<<gemmBlocksHid, 256, 0, stream>>>(agg, bufA, Wl1, Wr1, b1, bufB, NODES);
    k_zero<<<1, 2 * HID, 0, stream>>>(stat, 2 * HID);
    k_bn_stats<<<bnBlocks, HID, 0, stream>>>(bufB, stat, NODES);
    k_bn_relu<<<(int)((nagg + zb - 1) / zb), zb, 0, stream>>>(bufB, stat, g1, be1, (int)nagg);

    // ---- layer 2: SAGE(H->OUT)  -> d_out ----
    k_zero<<<(int)((nagg + zb - 1) / zb), zb, 0, stream>>>(agg, (int)nagg);
    k_edge_agg<<<(int)((eth + zb - 1) / zb), zb, 0, stream>>>(bufB, src, dst, agg, EDGES);
    k_divdeg<<<(int)((nagg + zb - 1) / zb), zb, 0, stream>>>(agg, deg, (int)nagg);
    k_sage_gemm<DOUTC><<<gemmBlocksOut, 256, 0, stream>>>(agg, bufB, Wl2, Wr2, b2, out, NODES);
}
